// CDConv_Model_73933567033946
// MI455X (gfx1250) — compile-verified, matrix operand order checked by reading, not run on hardware
//
#include <hip/hip_runtime.h>

// ---------------------------------------------------------------------------
// CDConv model for MI455X (gfx1250, wave32, WMMA).
// Heavy GEMMs use v_wmma_f32_16x16x32_bf16 (bf16 A/B staged in LDS, f32 acc).
// Tile staging is vectorized: global_load_b128 -> regs -> ds_store_b128.
// Geometry / kernel-MLP / stencil aggregation stay in f32 VALU (~1% of FLOPs).
// ---------------------------------------------------------------------------

constexpr int NPTS = 8192;
constexpr int LWIN = 11;
constexpr int SHALF = 5;
constexpr int KKER = 24;

typedef __attribute__((ext_vector_type(16))) __bf16 v16bf;
typedef __attribute__((ext_vector_type(8)))  __bf16 bf16x8;
typedef __attribute__((ext_vector_type(8)))  float  v8f;
typedef __attribute__((ext_vector_type(4)))  float  f32x4;

__device__ __forceinline__ float leaky(float v, float s) { return v > 0.f ? v : v * s; }

// ---------------- position sum (single workgroup reduction) ----------------
__global__ void pos_sum_kernel(const float* __restrict__ pos, float* __restrict__ sum3) {
  __shared__ float rx[256], ry[256], rz[256];
  float ax = 0.f, ay = 0.f, az = 0.f;
  for (int n = threadIdx.x; n < NPTS; n += 256) {
    ax += pos[n * 3 + 0]; ay += pos[n * 3 + 1]; az += pos[n * 3 + 2];
  }
  rx[threadIdx.x] = ax; ry[threadIdx.x] = ay; rz[threadIdx.x] = az;
  __syncthreads();
  for (int s = 128; s > 0; s >>= 1) {
    if ((int)threadIdx.x < s) {
      rx[threadIdx.x] += rx[threadIdx.x + s];
      ry[threadIdx.x] += ry[threadIdx.x + s];
      rz[threadIdx.x] += rz[threadIdx.x + s];
    }
    __syncthreads();
  }
  if (threadIdx.x == 0) { sum3[0] = rx[0]; sum3[1] = ry[0]; sum3[2] = rz[0]; }
}

__global__ void center_kernel(const float* __restrict__ pos, const float* __restrict__ sum3,
                              float* __restrict__ posc) {
  int i = blockIdx.x * 256 + threadIdx.x;
  if (i < NPTS * 3) posc[i] = pos[i] - sum3[i % 3];
}

// ---------------- orientation frames ----------------
__device__ __forceinline__ void norm3(float& x, float& y, float& z) {
  float n = sqrtf(x * x + y * y + z * z);
  float inv = 1.f / fmaxf(n, 1e-12f);
  x *= inv; y *= inv; z *= inv;
}

__global__ void ori_kernel(const float* __restrict__ pos, float* __restrict__ ori) {
  int n = blockIdx.x * 256 + threadIdx.x;
  if (n >= NPTS) return;
  int m = min(max(n, 1), NPTS - 2);
  float u2x = pos[m * 3 + 0] - pos[(m - 1) * 3 + 0];
  float u2y = pos[m * 3 + 1] - pos[(m - 1) * 3 + 1];
  float u2z = pos[m * 3 + 2] - pos[(m - 1) * 3 + 2];
  norm3(u2x, u2y, u2z);
  float u1x = pos[(m + 1) * 3 + 0] - pos[m * 3 + 0];
  float u1y = pos[(m + 1) * 3 + 1] - pos[m * 3 + 1];
  float u1z = pos[(m + 1) * 3 + 2] - pos[m * 3 + 2];
  norm3(u1x, u1y, u1z);
  float bx = u2x - u1x, by = u2y - u1y, bz = u2z - u1z;
  norm3(bx, by, bz);
  float nx = u2y * u1z - u2z * u1y;
  float ny = u2z * u1x - u2x * u1z;
  float nz = u2x * u1y - u2y * u1x;
  norm3(nx, ny, nz);
  float ox = by * nz - bz * ny;
  float oy = bz * nx - bx * nz;
  float oz = bx * ny - by * nx;
  norm3(ox, oy, oz);
  float* o = ori + (size_t)n * 9;
  o[0] = bx; o[1] = by; o[2] = bz;
  o[3] = nx; o[4] = ny; o[5] = nz;
  o[6] = ox; o[7] = oy; o[8] = oz;
}

// ---------------- shared geometry features: [local(3), orirel(3), dist] ----
__global__ void geom_kernel(const float* __restrict__ pos, const float* __restrict__ ori,
                            float* __restrict__ geom) {
  int idx = blockIdx.x * 256 + threadIdx.x;
  if (idx >= NPTS * LWIN) return;
  int n = idx / LWIN, l = idx % LWIN;
  int j = n + l - SHALF;
  int jc = min(max(j, 0), NPTS - 1);
  float dx = pos[jc * 3 + 0] - pos[n * 3 + 0];
  float dy = pos[jc * 3 + 1] - pos[n * 3 + 1];
  float dz = pos[jc * 3 + 2] - pos[n * 3 + 2];
  float dist = sqrtf(dx * dx + dy * dy + dz * dz);
  float inv = 1.f / (dist + 1e-9f);
  float ux = dx * inv, uy = dy * inv, uz = dz * inv;
  const float* on = ori + (size_t)n * 9;
  const float* oj = ori + (size_t)jc * 9;
  float* g = geom + (size_t)idx * 7;
  g[0] = on[0] * ux + on[1] * uy + on[2] * uz;
  g[1] = on[3] * ux + on[4] * uy + on[5] * uz;
  g[2] = on[6] * ux + on[7] * uy + on[8] * uz;
  g[3] = on[0] * oj[0] + on[1] * oj[1] + on[2] * oj[2];
  g[4] = on[3] * oj[3] + on[4] * oj[4] + on[5] * oj[5];
  g[5] = on[6] * oj[6] + on[7] * oj[7] + on[8] * oj[8];
  g[6] = dist;
}

// ---------------- per-block kernel MLP: leaky(delta@W0 + b0, 0.2)*mask -----
__global__ void compute_ker_kernel(const float* __restrict__ geom,
                                   const float* __restrict__ W0,
                                   const float* __restrict__ b0,
                                   float r, float* __restrict__ ker) {
  int idx = blockIdx.x * 256 + threadIdx.x;
  if (idx >= NPTS * LWIN) return;
  int n = idx / LWIN, l = idx % LWIN;
  int j = n + l - SHALF;
  bool valid = (j >= 0) && (j < NPTS);
  const float* g = geom + (size_t)idx * 7;
  float f0 = g[0], f1 = g[1], f2 = g[2], f3 = g[3], f4 = g[4], f5 = g[5];
  float dist = g[6];
  float m = (valid && (dist < r)) ? 1.f : 0.f;
  float dr = dist / r;
  const float* w0l = W0 + (size_t)l * 7 * KKER;
  const float* b0l = b0 + (size_t)l * KKER;
  float* out = ker + (size_t)idx * KKER;
#pragma unroll 4
  for (int k = 0; k < KKER; ++k) {
    float acc = b0l[k];
    acc += f0 * w0l[0 * KKER + k];
    acc += f1 * w0l[1 * KKER + k];
    acc += f2 * w0l[2 * KKER + k];
    acc += f3 * w0l[3 * KKER + k];
    acc += f4 * w0l[4 * KKER + k];
    acc += f5 * w0l[5 * KKER + k];
    acc += dr * w0l[6 * KKER + k];
    out[k] = leaky(acc, 0.2f) * m;
  }
}

// ---------------- embedding gather ----------------
__global__ void embed_kernel(const int* __restrict__ S, const float* __restrict__ emb,
                             float* __restrict__ x) {
  int i = blockIdx.x * 256 + threadIdx.x;
  if (i < NPTS * 128) {
    int n = i >> 7, c = i & 127;
    x[i] = emb[S[n] * 128 + c];
  }
}

// ---------------- weight f32 -> bf16 transpose (Bt[n][k] = W[k][n]) --------
__global__ void cvt_w_kernel(const float* __restrict__ W, __bf16* __restrict__ Bt,
                             int Kc, int Nc) {
  int idx = blockIdx.x * 256 + threadIdx.x;
  if (idx < Kc * Nc) {
    int k = idx / Nc, n = idx % Nc;
    Bt[(size_t)n * Kc + k] = (__bf16)W[idx];
  }
}

// ---------------- stencil aggregation: agg[n,k,c] = sum_l ker*h -----------
__global__ __launch_bounds__(128)
void agg_kernel(const float* __restrict__ ker, const float* __restrict__ h,
                __bf16* __restrict__ agg, int w) {
  __shared__ float lk[LWIN * KKER];
  __shared__ float lh[LWIN * 256];
  int n = blockIdx.x;
  for (int i = threadIdx.x; i < LWIN * KKER; i += 128)
    lk[i] = ker[(size_t)n * LWIN * KKER + i];
  for (int l = 0; l < LWIN; ++l) {
    int j = min(max(n + l - SHALF, 0), NPTS - 1);
    for (int c = threadIdx.x; c < w; c += 128)
      lh[l * w + c] = h[(size_t)j * w + c];
  }
  __syncthreads();
  for (int k = 0; k < KKER; ++k) {
    for (int c = threadIdx.x; c < w; c += 128) {
      float acc = 0.f;
#pragma unroll
      for (int l = 0; l < LWIN; ++l) acc += lk[l * KKER + k] * lh[l * w + c];
      agg[((size_t)n * KKER + k) * w + c] = (__bf16)acc;
    }
  }
}

// ---------------- tiled bf16 WMMA GEMM -------------------------------------
// C[M x Nc] = epilogue( pre(A[M x Kc]) @ Bt^T ), Bt stored [Nc x Kc] bf16.
// Tile: 128 rows x 32 cols, 4 waves, each wave 32x32 via 2x2 v_wmma 16x16x32.
// Staging: 16B global loads -> registers -> 16B LDS stores (batched).
template <bool ABF, bool HASADD, bool HASBIAS>
__global__ __launch_bounds__(128)
void gemm_kernel(const void* __restrict__ Av, const __bf16* __restrict__ Bt,
                 float* __restrict__ C, int M, int Kc, int Nc,
                 float pre_slope, float out_slope,
                 const float* __restrict__ addsrc, const float* __restrict__ bias) {
  (void)M; (void)pre_slope;
  __shared__ __attribute__((aligned(16))) __bf16 lA[128][40];  // 80B stride: 16B-aligned chunks, conflict-free
  __shared__ __attribute__((aligned(16))) __bf16 lB[32][40];
  const int tid = threadIdx.x;
  const int m0 = blockIdx.x * 128;
  const int n0 = blockIdx.y * 32;
  const int lane = tid & 31;
  const int wave = tid >> 5;
  const int l15 = lane & 15;
  const int kb = (lane >> 4) * 16;  // K-half select: lane half h holds contiguous k = 16h..16h+15 for both A and B
  const float* Af = (const float*)Av;
  const __bf16* Ab = (const __bf16*)Av;
  // per-thread staging chunk coordinates (8 elements along K per chunk)
  const int br = tid >> 2;          // B chunk row (output col), 0..31
  const int bc8 = (tid & 3) * 8;    // B chunk K offset
  v8f c00 = {}, c01 = {}, c10 = {}, c11 = {};

  for (int k0 = 0; k0 < Kc; k0 += 32) {
    // ---- global -> registers (all loads issued before any LDS store) ----
    bf16x8 va[4];
#pragma unroll
    for (int i = 0; i < 4; ++i) {
      int idx = tid + i * 128;
      int r = idx >> 2, c8 = (idx & 3) * 8;
      const size_t base = (size_t)(m0 + r) * Kc + k0 + c8;
      if constexpr (ABF) {
        va[i] = *(const bf16x8*)(Ab + base);
      } else {
        f32x4 f0 = *(const f32x4*)(Af + base);
        f32x4 f1 = *(const f32x4*)(Af + base + 4);
        bf16x8 v;
#pragma unroll
        for (int t = 0; t < 4; ++t) {
          v[t] = (__bf16)leaky(f0[t], pre_slope);
          v[4 + t] = (__bf16)leaky(f1[t], pre_slope);
        }
        va[i] = v;
      }
    }
    bf16x8 vb = *(const bf16x8*)(Bt + (size_t)(n0 + br) * Kc + k0 + bc8);

    __syncthreads();  // previous iteration's fragment reads complete
    // ---- registers -> LDS (16B stores) ----
#pragma unroll
    for (int i = 0; i < 4; ++i) {
      int idx = tid + i * 128;
      int r = idx >> 2, c8 = (idx & 3) * 8;
      *(bf16x8*)&lA[r][c8] = va[i];
    }
    *(bf16x8*)&lB[br][bc8] = vb;
    __syncthreads();

    // ---- fragments + WMMA ----
    v16bf a0, a1, b0, b1;
#pragma unroll
    for (int i = 0; i < 16; ++i) {
      a0[i] = lA[wave * 32 + l15][kb + i];
      a1[i] = lA[wave * 32 + 16 + l15][kb + i];
      b0[i] = lB[l15][kb + i];
      b1[i] = lB[16 + l15][kb + i];
    }
    c00 = __builtin_amdgcn_wmma_f32_16x16x32_bf16(false, a0, false, b0, (short)0, c00, false, false);
    c01 = __builtin_amdgcn_wmma_f32_16x16x32_bf16(false, a0, false, b1, (short)0, c01, false, false);
    c10 = __builtin_amdgcn_wmma_f32_16x16x32_bf16(false, a1, false, b0, (short)0, c10, false, false);
    c11 = __builtin_amdgcn_wmma_f32_16x16x32_bf16(false, a1, false, b1, (short)0, c11, false, false);
  }

  // epilogue: C/D layout -> lanes 0-15: rows v, col=lane; lanes 16-31: rows 8+v
  const int rsel = (lane >> 4) * 8;
  auto emit = [&](float val, int row, int col) {
    if constexpr (HASBIAS) val += bias[col];
    if constexpr (HASADD) val += addsrc[(size_t)row * Nc + col];
    val = val > 0.f ? val : val * out_slope;
    C[(size_t)row * Nc + col] = val;
  };
#pragma unroll
  for (int v = 0; v < 8; ++v) {
    int r0 = m0 + wave * 32 + rsel + v;
    int r1 = r0 + 16;
    int cA = n0 + l15;
    int cB = cA + 16;
    emit(c00[v], r0, cA);
    emit(c01[v], r0, cB);
    emit(c10[v], r1, cA);
    emit(c11[v], r1, cB);
  }
}

// ---------------- classifier head: leaky(h1)@Wc2 + bc2, log_softmax(26) ---
__global__ __launch_bounds__(32)
void final_kernel(const float* __restrict__ h1, const float* __restrict__ Wc2,
                  const float* __restrict__ bc2, float* __restrict__ out) {
  __shared__ float lg[26];
  __shared__ float stats[2];
  int n = blockIdx.x;
  int j = threadIdx.x;
  if (j < 26) {
    float acc = bc2[j];
    for (int m = 0; m < 1024; ++m) {
      float a = h1[(size_t)n * 1024 + m];
      a = leaky(a, 0.2f);
      acc += a * Wc2[m * 26 + j];
    }
    lg[j] = acc;
  }
  __syncthreads();
  if (j == 0) {
    float mx = lg[0];
    for (int t = 1; t < 26; ++t) mx = fmaxf(mx, lg[t]);
    float s = 0.f;
    for (int t = 0; t < 26; ++t) s += expf(lg[t] - mx);
    stats[0] = mx;
    stats[1] = logf(s);
  }
  __syncthreads();
  if (j < 26) out[(size_t)n * 26 + j] = lg[j] - stats[0] - stats[1];
}

// ---------------------------------------------------------------------------
extern "C" void kernel_launch(void* const* d_in, const int* in_sizes, int n_in,
                              void* d_out, int out_size, void* d_ws, size_t ws_size,
                              hipStream_t stream) {
  (void)in_sizes; (void)n_in; (void)out_size; (void)ws_size;

  // --- parse inputs (setup_inputs dict order; blocks flatten in insertion order)
  int ii = 0;
  const int*   S   = (const int*)d_in[ii++];
  const float* pos = (const float*)d_in[ii++];
  ii++;  // mask (unused by reference)
  const float* emb = (const float*)d_in[ii++];

  const int   in_chs[8] = {128, 128, 128, 256, 256, 512, 512, 1024};
  const int   cs[8]     = {128, 128, 256, 256, 512, 512, 1024, 1024};
  const float rads[8]   = {4.f, 4.f, 6.f, 6.f, 8.f, 8.f, 10.f, 10.f};
  struct Blk { const float *W_id, *W_in, *W0, *b0, *Wc, *W_out; };
  Blk blk[8];
  for (int b = 0; b < 8; ++b) {
    blk[b].W_id = nullptr;
    if (in_chs[b] != cs[b]) blk[b].W_id = (const float*)d_in[ii++];
    blk[b].W_in  = (const float*)d_in[ii++];
    blk[b].W0    = (const float*)d_in[ii++];
    blk[b].b0    = (const float*)d_in[ii++];
    blk[b].Wc    = (const float*)d_in[ii++];
    blk[b].W_out = (const float*)d_in[ii++];
  }
  const float* Wc1 = (const float*)d_in[ii++];
  const float* bc1 = (const float*)d_in[ii++];
  const float* Wc2 = (const float*)d_in[ii++];
  const float* bc2 = (const float*)d_in[ii++];

  // --- workspace bump allocator
  char* p = (char*)d_ws;
  auto alloc = [&](size_t bytes) -> char* {
    char* r = p;
    p += (bytes + 255) & ~(size_t)255;
    return r;
  };
  float*  sum3   = (float*)alloc(16);
  float*  posc   = (float*)alloc((size_t)NPTS * 3 * 4);
  float*  ori    = (float*)alloc((size_t)NPTS * 9 * 4);
  float*  geom   = (float*)alloc((size_t)NPTS * LWIN * 7 * 4);
  float*  ker    = (float*)alloc((size_t)NPTS * LWIN * KKER * 4);
  float*  xA     = (float*)alloc((size_t)NPTS * 1024 * 4);
  float*  xB     = (float*)alloc((size_t)NPTS * 1024 * 4);
  float*  hbuf   = (float*)alloc((size_t)NPTS * 256 * 4);
  float*  hbuf2  = (float*)alloc((size_t)NPTS * 256 * 4);
  __bf16* aggbuf = (__bf16*)alloc((size_t)NPTS * KKER * 256 * 2);
  __bf16* wbf    = (__bf16*)alloc((size_t)6144 * 256 * 2);

  auto cvt = [&](const float* W, int Kc, int Nc) {
    int tot = Kc * Nc;
    cvt_w_kernel<<<(tot + 255) / 256, 256, 0, stream>>>(W, wbf, Kc, Nc);
  };

  // --- geometry (block-independent)
  pos_sum_kernel<<<1, 256, 0, stream>>>(pos, sum3);
  center_kernel<<<(NPTS * 3 + 255) / 256, 256, 0, stream>>>(pos, sum3, posc);
  ori_kernel<<<(NPTS + 255) / 256, 256, 0, stream>>>(posc, ori);
  geom_kernel<<<(NPTS * LWIN + 255) / 256, 256, 0, stream>>>(posc, ori, geom);
  embed_kernel<<<(NPTS * 128 + 255) / 256, 256, 0, stream>>>(S, emb, xA);

  // --- residual blocks
  float* cur = xA;
  float* alt = xB;
  for (int b = 0; b < 8; ++b) {
    int in_ch = in_chs[b], c = cs[b], w = c / 4;
    compute_ker_kernel<<<(NPTS * LWIN + 255) / 256, 256, 0, stream>>>(
        geom, blk[b].W0, blk[b].b0, rads[b], ker);

    float* ident = cur;
    if (blk[b].W_id) {  // ident = leaky(x,0.1) @ W_id
      cvt(blk[b].W_id, in_ch, c);
      gemm_kernel<false, false, false><<<dim3(NPTS / 128, c / 32), 128, 0, stream>>>(
          cur, wbf, alt, NPTS, in_ch, c, 0.1f, 1.0f, nullptr, nullptr);
      ident = alt;
    }
    // h = leaky(leaky(x,0.2) @ W_in, 0.2)
    cvt(blk[b].W_in, in_ch, w);
    gemm_kernel<false, false, false><<<dim3(NPTS / 128, w / 32), 128, 0, stream>>>(
        cur, wbf, hbuf, NPTS, in_ch, w, 0.2f, 0.2f, nullptr, nullptr);
    // agg[n, k*w + c'] = sum_l ker[n,l,k] * h[n+l-5, c']
    agg_kernel<<<NPTS, 128, 0, stream>>>(ker, hbuf, aggbuf, w);
    // h_conv = agg @ Wc
    cvt(blk[b].Wc, KKER * w, w);
    gemm_kernel<true, false, false><<<dim3(NPTS / 128, w / 32), 128, 0, stream>>>(
        aggbuf, wbf, hbuf2, NPTS, KKER * w, w, 1.0f, 1.0f, nullptr, nullptr);
    // x' = leaky(h_conv,0.1) @ W_out + ident
    cvt(blk[b].W_out, w, c);
    float* dst = blk[b].W_id ? cur : alt;
    gemm_kernel<false, true, false><<<dim3(NPTS / 128, c / 32), 128, 0, stream>>>(
        hbuf2, wbf, dst, NPTS, w, c, 0.1f, 1.0f, ident, nullptr);
    if (!blk[b].W_id) { float* t = cur; cur = alt; alt = t; }
  }

  // --- classifier: h1 = leaky(x,0.2) @ Wc1 + bc1 ; head + log_softmax
  cvt(Wc1, 1024, 1024);
  gemm_kernel<false, false, true><<<dim3(NPTS / 128, 1024 / 32), 128, 0, stream>>>(
      cur, wbf, alt, NPTS, 1024, 1024, 0.2f, 1.0f, nullptr, bc1);
  final_kernel<<<NPTS, 32, 0, stream>>>(alt, Wc2, bc2, (float*)d_out);
}
